// NormalizedRegularizedGCNLayer_72301479461276
// MI455X (gfx1250) — compile-verified, matrix operand order checked by reading, not run on hardware
//
#include <hip/hip_runtime.h>
#include <hip/hip_bf16.h>

#define N_IN_DIM 128
#define N_OUT_DIM 128
#define BN_EPS 1e-5f
#define FILL_VALUE 1.0f

typedef __attribute__((ext_vector_type(2)))  float    v2f;
typedef __attribute__((ext_vector_type(8)))  float    v8f;
typedef __attribute__((ext_vector_type(16))) _Float16 v16h;

#if defined(__has_builtin)
#if __has_builtin(__builtin_amdgcn_wmma_f32_16x16x4_f32)
#define USE_F32_WMMA 1
#endif
#endif

// ---------------------------------------------------------------------------
// K0: initialize workspace: deg = FILL_VALUE (self-loop weight), ACC = 0,
//     column-stat accumulators (256 floats) = 0.
// ---------------------------------------------------------------------------
__global__ void k_init(float* __restrict__ deg, float* __restrict__ acc,
                       float* __restrict__ stats, int N, long long total) {
  long long i = (long long)blockIdx.x * blockDim.x + threadIdx.x;
  if (i < total) acc[i] = 0.0f;
  if (i < (long long)N) deg[i] = FILL_VALUE;
  if (i < 2 * N_OUT_DIM) stats[i] = 0.0f;
}

// ---------------------------------------------------------------------------
// K1: degree accumulation: deg[col[e]] += w[e]
// ---------------------------------------------------------------------------
__global__ void k_degree(const int* __restrict__ col, const float* __restrict__ w,
                         float* __restrict__ deg, int E) {
  int e = blockIdx.x * blockDim.x + threadIdx.x;
  if (e < E) atomicAdd(&deg[col[e]], w[e]);
}

// ---------------------------------------------------------------------------
// K2: dinv = deg > 0 ? rsqrt(deg) : 0
// ---------------------------------------------------------------------------
__global__ void k_dinv(const float* __restrict__ deg, float* __restrict__ dinv, int N) {
  int i = blockIdx.x * blockDim.x + threadIdx.x;
  if (i < N) {
    float d = deg[i];
    dinv[i] = (d > 0.0f) ? rsqrtf(d) : 0.0f;
  }
}

// ---------------------------------------------------------------------------
// K3: X = A @ W  via CDNA5 WMMA (V_WMMA_F32_16X16X4_F32, fp32 precision).
//     One wave computes one 16x16 output tile; K=128 -> 32 chained WMMAs.
//     A: [nRows, 128] row-major;  W: [128, 128] row-major;  X: [nRows, 128].
// ---------------------------------------------------------------------------
__global__ __launch_bounds__(128)
void k_gemm_wmma(const float* __restrict__ A, const float* __restrict__ Wm,
                 float* __restrict__ X, int nRows) {
  const int lane   = threadIdx.x & 31;
  const int wave   = blockIdx.x * 4 + (threadIdx.x >> 5);
  const int tilesN = N_OUT_DIM / 16;          // 8
  const int tileM  = wave / tilesN;
  const int tileN  = wave % tilesN;
  if (tileM * 16 >= nRows) return;

  const int mn = lane & 15;                   // row of A tile / col of B tile
  const int hi = lane >> 4;                   // lane half selects K sub-range
  v8f c = {};

#if defined(USE_F32_WMMA)
  // f32 A 16x4 layout: lanes 0-15 -> K = k+0,k+1 ; lanes 16-31 -> K = k+2,k+3
  const int kh = hi * 2;
  const float* arow = A  + (size_t)(tileM * 16 + mn) * N_IN_DIM;
  const float* bcol = Wm + tileN * 16 + mn;   // column (tileN*16+mn) of W
  #pragma unroll
  for (int k = 0; k < N_IN_DIM; k += 4) {
    v2f a, b;
    a.x = arow[k + kh];
    a.y = arow[k + kh + 1];
    b.x = bcol[(size_t)(k + kh)     * N_OUT_DIM];
    b.y = bcol[(size_t)(k + kh + 1) * N_OUT_DIM];
    c = __builtin_amdgcn_wmma_f32_16x16x4_f32(
        /*neg_a=*/false, a, /*neg_b=*/false, b,
        /*c_mod=*/(short)0, c, /*reuse_a=*/false, /*reuse_b=*/false);
  }
#else
  // Fallback: convert to f16, K=32 per WMMA (codegen-confirmed builtin).
  const int kbase = hi * 8;
  const float* arow = A  + (size_t)(tileM * 16 + mn) * N_IN_DIM;
  const float* bcol = Wm + tileN * 16 + mn;
  for (int k = 0; k < N_IN_DIM; k += 32) {
    v16h a, b;
    #pragma unroll
    for (int j = 0; j < 16; ++j) {
      int kk = k + kbase + j + ((j >= 8) ? 8 : 0);
      a[j] = (_Float16)arow[kk];
      b[j] = (_Float16)bcol[(size_t)kk * N_OUT_DIM];
    }
    c = __builtin_amdgcn_wmma_f32_16x16x32_f16(
        false, a, false, b, (short)0, c, false, false);
  }
#endif

  // C/D layout: VGPR r holds M = r + 8*hi, N = lane%16.
  // Full-tile fast path: wave-uniform bound check -> unconditional stores,
  // no per-element exec-mask branching in the epilogue.
  const int nOut  = tileN * 16 + mn;
  const int mBase = tileM * 16 + hi * 8;
  float* dst = X + (size_t)mBase * N_OUT_DIM + nOut;
  if (tileM * 16 + 16 <= nRows) {
    #pragma unroll
    for (int r = 0; r < 8; ++r) {
      dst[(size_t)r * N_OUT_DIM] = c[r];
    }
  } else {
    #pragma unroll
    for (int r = 0; r < 8; ++r) {
      if (mBase + r < nRows) dst[(size_t)r * N_OUT_DIM] = c[r];
    }
  }
}

// ---------------------------------------------------------------------------
// K4: edge message scatter.  One wave per edge; each lane handles 4 dims.
//     ACC[col] += X[row] * (dinv[row] * w * dinv[col])
// ---------------------------------------------------------------------------
__global__ __launch_bounds__(256)
void k_scatter(const int* __restrict__ row, const int* __restrict__ col,
               const float* __restrict__ w, const float* __restrict__ dinv,
               const float* __restrict__ X, float* __restrict__ ACC, int E) {
  const int lane = threadIdx.x & 31;
  const int e = blockIdx.x * 8 + (threadIdx.x >> 5);
  if (e >= E) return;
  const int r = row[e];
  const int cgt = col[e];
  const float nrm = dinv[r] * w[e] * dinv[cgt];
  const float4 v = ((const float4*)(X + (size_t)r * N_OUT_DIM))[lane];
  float* dst = ACC + (size_t)cgt * N_OUT_DIM + lane * 4;
  atomicAdd(dst + 0, v.x * nrm);
  atomicAdd(dst + 1, v.y * nrm);
  atomicAdd(dst + 2, v.z * nrm);
  atomicAdd(dst + 3, v.w * nrm);
}

// ---------------------------------------------------------------------------
// K5: fused self-loop + bias + ReLU, in-place into ACC, and column statistics
//     (sum, sum-of-squares) via per-thread register accumulation + atomics.
//     Thread d owns column d (blockDim = 128).
// ---------------------------------------------------------------------------
__global__ __launch_bounds__(128)
void k_finalize_stats(const float* __restrict__ X, const float* __restrict__ dinv,
                      const float* __restrict__ b, float* __restrict__ ACC,
                      float* __restrict__ colsum, float* __restrict__ colsq,
                      int N, int rowsPerBlock) {
  const int d = threadIdx.x;
  const float bb = b[d];
  const int r0 = blockIdx.x * rowsPerBlock;
  const int r1 = min(N, r0 + rowsPerBlock);
  float s = 0.0f, s2 = 0.0f;
  for (int i = r0; i < r1; ++i) {
    const float di = dinv[i];
    const float selfnorm = di * di * FILL_VALUE;           // dinv * w_self * dinv
    float h = ACC[(size_t)i * N_OUT_DIM + d]
            + X[(size_t)i * N_OUT_DIM + d] * selfnorm + bb;
    h = fmaxf(h, 0.0f);
    ACC[(size_t)i * N_OUT_DIM + d] = h;
    s += h;
    s2 += h * h;
  }
  atomicAdd(&colsum[d], s);
  atomicAdd(&colsq[d], s2);
}

// ---------------------------------------------------------------------------
// K6: BN parameters from column sums (1 block of 128 threads).
//     scale = gamma * rsqrt(var + eps);  shift = beta - mean * scale
// ---------------------------------------------------------------------------
__global__ __launch_bounds__(128)
void k_bn_params(const float* __restrict__ colsum, const float* __restrict__ colsq,
                 const float* __restrict__ gamma, const float* __restrict__ beta,
                 float* __restrict__ scale, float* __restrict__ shift, int N) {
  const int d = threadIdx.x;
  const float invN = 1.0f / (float)N;
  const float mean = colsum[d] * invN;
  float var = colsq[d] * invN - mean * mean;
  var = fmaxf(var, 0.0f);
  const float sc = gamma[d] * rsqrtf(var + BN_EPS);
  scale[d] = sc;
  shift[d] = beta[d] - mean * sc;
}

// ---------------------------------------------------------------------------
// K7: final affine: out = h * scale + shift
// ---------------------------------------------------------------------------
__global__ __launch_bounds__(256)
void k_bn_apply(const float* __restrict__ ACC, const float* __restrict__ scale,
                const float* __restrict__ shift, float* __restrict__ out,
                long long total) {
  long long i = (long long)blockIdx.x * blockDim.x + threadIdx.x;
  if (i < total) {
    int d = (int)(i & (N_OUT_DIM - 1));
    out[i] = fmaf(ACC[i], scale[d], shift[d]);
  }
}

// ---------------------------------------------------------------------------
extern "C" void kernel_launch(void* const* d_in, const int* in_sizes, int n_in,
                              void* d_out, int out_size, void* d_ws, size_t ws_size,
                              hipStream_t stream) {
  const float* feat  = (const float*)d_in[0];   // [N, 128]
  const int*   eidx  = (const int*)d_in[1];     // [2, E] flat: rows then cols
  const float* ew    = (const float*)d_in[2];   // [E]
  const float* Wm    = (const float*)d_in[3];   // [128, 128]
  const float* bvec  = (const float*)d_in[4];   // [128]
  const float* gamma = (const float*)d_in[5];   // [128]
  const float* beta  = (const float*)d_in[6];   // [128]

  const int N = in_sizes[0] / N_IN_DIM;
  const int E = in_sizes[1] / 2;
  const int* rowIdx = eidx;
  const int* colIdx = eidx + E;

  // Workspace layout (floats): deg[N] | dinv[N] | X[N*128] | ACC[N*128] |
  //                            colsum[128] | colsq[128] | scale[128] | shift[128]
  float* ws     = (float*)d_ws;
  float* deg    = ws;
  float* dinv   = ws + (size_t)N;
  float* X      = ws + 2 * (size_t)N;
  float* ACC    = X + (size_t)N * N_OUT_DIM;
  float* colsum = ACC + (size_t)N * N_OUT_DIM;
  float* colsq  = colsum + N_OUT_DIM;
  float* scale  = colsq + N_OUT_DIM;
  float* shift  = scale + N_OUT_DIM;

  const long long total = (long long)N * N_OUT_DIM;

  // K0: init
  {
    int blocks = (int)((total + 255) / 256);
    k_init<<<blocks, 256, 0, stream>>>(deg, ACC, colsum, N, total);
  }
  // K1: degree
  k_degree<<<(E + 255) / 256, 256, 0, stream>>>(colIdx, ew, deg, E);
  // K2: dinv
  k_dinv<<<(N + 255) / 256, 256, 0, stream>>>(deg, dinv, N);
  // K3: GEMM via WMMA (one wave per 16x16 tile, 4 waves/block)
  {
    int tiles = ((N + 15) / 16) * (N_OUT_DIM / 16);
    k_gemm_wmma<<<(tiles + 3) / 4, 128, 0, stream>>>(feat, Wm, X, N);
  }
  // K4: edge scatter (8 edges / 256-thread block)
  k_scatter<<<(E + 7) / 8, 256, 0, stream>>>(rowIdx, colIdx, ew, dinv, X, ACC, E);
  // K5: self-loop + bias + relu + column stats
  {
    int blocks = 1024;
    int rpb = (N + blocks - 1) / blocks;
    k_finalize_stats<<<blocks, 128, 0, stream>>>(X, dinv, bvec, ACC, colsum, colsq, N, rpb);
  }
  // K6: BN parameters
  k_bn_params<<<1, 128, 0, stream>>>(colsum, colsq, gamma, beta, scale, shift, N);
  // K7: apply BN -> d_out
  {
    int blocks = (int)((total + 255) / 256);
    k_bn_apply<<<blocks, 256, 0, stream>>>(ACC, scale, shift, (float*)d_out, total);
  }
}
